// CANCRB_Down_2757369004649
// MI455X (gfx1250) — compile-verified, matrix operand order checked by reading, not run on hardware
//
#include <hip/hip_runtime.h>

typedef __attribute__((ext_vector_type(16))) __bf16 v16bf;
typedef __attribute__((ext_vector_type(8)))  __bf16 v8bf;
typedef __attribute__((ext_vector_type(8)))  float  v8f;
typedef __attribute__((ext_vector_type(4)))  unsigned u32x4;
typedef __attribute__((ext_vector_type(8)))  unsigned u32x8;

namespace {
constexpr int B  = 8;
constexpr int C  = 64;
constexpr int H  = 128;
constexpr int W  = 128;
constexpr int Nn = H * W;      // 16384 pixels
constexpr int S  = 32;         // clusters
constexpr int K9 = 9;
constexpr int CK = C * K9;     // 576
constexpr int O  = 64;
constexpr int KM_ITERS = 8;
constexpr int KSTEPS = CK / 32;                   // 18 WMMA k-steps
constexpr int W2P_ELEMS = 4 * KSTEPS * 32 * 16;   // packed w2: [otile][ks][lane][i]
}

// K index for element i of a 16-bit WMMA A/B operand (ISA 7.12.2, 16-bit 16x32):
// i=0..7 -> K = laneHi*8 + i ; i=8..15 -> K = 16 + laneHi*8 + (i-8)
__device__ __forceinline__ int kmap(int i, int laneHi) {
  return ((i >> 3) << 4) + (laneHi << 3) + (i & 7);
}

// Assemble a v16bf A/B fragment from two contiguous 16B chunks of a row-major K-line.
__device__ __forceinline__ v16bf frag16(const __bf16* row, int kbase, int laneHi) {
  v8bf lo = *(const v8bf*)(row + kbase + laneHi * 8);
  v8bf hi = *(const v8bf*)(row + kbase + 16 + laneHi * 8);
  return __builtin_shufflevector(lo, hi, 0, 1, 2, 3, 4, 5, 6, 7,
                                         8, 9, 10, 11, 12, 13, 14, 15);
}

// ---------------------------------------------------------------- utilities
__global__ void zero_f32(float* __restrict__ p, int n) {
  int i = blockIdx.x * 256 + threadIdx.x;
  if (i < n) p[i] = 0.f;
}

__global__ void init_cent(const float* __restrict__ x, float* __restrict__ cent) {
  int t = blockIdx.x * 256 + threadIdx.x;           // B*S*C = 16384
  if (t >= B * S * C) return;
  int b = t >> 11;
  int r = t & 2047;
  int s = r >> 6;
  int c = r & 63;
  cent[t] = x[((size_t)(b * C + c)) * Nn + s * (Nn / S)];
}

// Pre-pack w2 (f32 [O][C*9]) into bf16 in the exact WMMA B-operand register layout.
__global__ void pack_w2(const float* __restrict__ w2, unsigned short* __restrict__ w2p) {
  int e = blockIdx.x * 256 + threadIdx.x;
  if (e >= W2P_ELEMS) return;
  int i     = e & 15;
  int lane  = (e >> 4) & 31;
  int ks    = (e >> 9) % KSTEPS;
  int otile = e / (KSTEPS * 512);
  int q = ks * 32 + kmap(i, lane >> 4);
  int o = otile * 16 + (lane & 15);
  __bf16 bv = (__bf16)w2[(size_t)o * CK + q];
  w2p[e] = __builtin_bit_cast(unsigned short, bv);
}

// ---------------------------------------------------------------- k-means assign (WMMA)
// grid = B * (Nn/128), block = 256 (8 waves, 16 pixels/wave, all 32 clusters)
__global__ void kmeans_assign(const float* __restrict__ x,
                              const float* __restrict__ cent,
                              int* __restrict__ idx) {
  __shared__ __bf16 centl[S * C];       // 4 KB, row-major [s][c]
  __shared__ __bf16 xl[128 * C];        // 16 KB, row-major [pixel][c]
  __shared__ float  cnorm[S];
  __shared__ float  dist[8][16][S];     // 16 KB

  const int b    = blockIdx.x >> 7;
  const int base = (blockIdx.x & 127) * 128;
  const int tid  = threadIdx.x;
  const int lane = tid & 31, wave = tid >> 5;
  const int lane16 = lane & 15, laneHi = lane >> 4;

  const float* cb = cent + (size_t)b * S * C;
  const float* xb = x + (size_t)b * C * Nn;
  for (int e = tid; e < S * C; e += 256) centl[e] = (__bf16)cb[e];
  for (int e = tid; e < 128 * C; e += 256) {       // [p][c] (transposed stage)
    int p = e >> 6, c = e & 63;
    xl[e] = (__bf16)xb[(size_t)c * Nn + base + p];
  }
  if (tid < S) {
    float s2 = 0.f;
    for (int c = 0; c < C; ++c) { float v = cb[tid * C + c]; s2 += v * v; }
    cnorm[tid] = s2;
  }
  __syncthreads();

  const int pixloc = wave * 16 + lane16;
  const __bf16* arow = xl + pixloc * C;
  const __bf16* b0r  = centl + lane16 * C;
  const __bf16* b1r  = centl + (16 + lane16) * C;

  v8f acc0 = {}, acc1 = {};
#pragma unroll
  for (int ks = 0; ks < 2; ++ks) {                 // K = 64 channels, 2 x 32
    v16bf a  = frag16(arow, ks * 32, laneHi);      // 2x ds_load_b128 each
    v16bf b0 = frag16(b0r,  ks * 32, laneHi);
    v16bf b1 = frag16(b1r,  ks * 32, laneHi);
    acc0 = __builtin_amdgcn_wmma_f32_16x16x32_bf16(false, a, false, b0, (short)0, acc0, false, false);
    acc1 = __builtin_amdgcn_wmma_f32_16x16x32_bf16(false, a, false, b1, (short)0, acc1, false, false);
  }
#pragma unroll
  for (int j = 0; j < 8; ++j) {
    int m = (laneHi << 3) + j;
    dist[wave][m][lane16]      = cnorm[lane16]      - 2.f * acc0[j];
    dist[wave][m][16 + lane16] = cnorm[16 + lane16] - 2.f * acc1[j];
  }
  __syncthreads();
  if (lane < 16) {                                 // argmin (first-min wins)
    float best = dist[wave][lane][0];
    int bi = 0;
    for (int s = 1; s < S; ++s) {
      float d = dist[wave][lane][s];
      if (d < best) { best = d; bi = s; }
    }
    idx[(size_t)b * Nn + base + wave * 16 + lane] = bi;
  }
}

// ---------------------------------------------------------------- centroid accumulate
__global__ void kmeans_update(const float* __restrict__ x, const int* __restrict__ idx,
                              float* __restrict__ cent_acc, float* __restrict__ cnt) {
  __shared__ float lacc[S * C];
  __shared__ float lcnt[S];
  const int b    = blockIdx.x >> 6;
  const int base = (blockIdx.x & 63) * 256;
  const int t    = threadIdx.x;

  for (int e = t; e < S * C; e += 256) lacc[e] = 0.f;
  if (t < S) lcnt[t] = 0.f;
  __syncthreads();

  const int n = base + t;
  const int s = idx[(size_t)b * Nn + n];
  atomicAdd(&lcnt[s], 1.f);
  const float* xb = x + (size_t)b * C * Nn;
#pragma unroll 1
  for (int c = 0; c < C; ++c) atomicAdd(&lacc[s * C + c], xb[(size_t)c * Nn + n]);
  __syncthreads();

  for (int e = t; e < S * C; e += 256) atomicAdd(&cent_acc[(size_t)b * S * C + e], lacc[e]);
  if (t < S) atomicAdd(&cnt[b * S + t], lcnt[t]);
}

__global__ void norm_cent(const float* __restrict__ acc, const float* __restrict__ cnt,
                          float* __restrict__ cent) {
  int t = blockIdx.x * 256 + threadIdx.x;
  if (t < B * S * C) cent[t] = acc[t] / fmaxf(cnt[t >> 6], 1.f);
}

// ---------------------------------------------------------------- cluster mean patches
__global__ void patch_acc(const float* __restrict__ x, const int* __restrict__ idx,
                          float* __restrict__ cent_p, float* __restrict__ cnt2) {
  const int b    = blockIdx.x >> 6;
  const int base = (blockIdx.x & 63) * 256;
  const int n    = base + threadIdx.x;
  const int s    = idx[(size_t)b * Nn + n];
  atomicAdd(&cnt2[b * S + s], 1.f);
  const int h = n >> 7, w = n & 127;
  const float* xb = x + (size_t)b * C * Nn;
  float* dst = cent_p + ((size_t)(b * S + s)) * CK;
#pragma unroll 1
  for (int c = 0; c < C; ++c) {
#pragma unroll 1
    for (int k = 0; k < K9; ++k) {
      int h2 = h + (k / 3) - 1, w2c = w + (k % 3) - 1;
      float v = 0.f;
      if ((unsigned)h2 < (unsigned)H && (unsigned)w2c < (unsigned)W)
        v = xb[(size_t)c * Nn + h2 * W + w2c];
      atomicAdd(&dst[c * K9 + k], v);
    }
  }
}

__global__ void norm_centp(float* __restrict__ cp, const float* __restrict__ cnt2) {
  int t = blockIdx.x * 256 + threadIdx.x;
  if (t < B * S * CK) cp[t] /= fmaxf(cnt2[t / CK], 1.f);
}

// ---------------------------------------------------------------- attention
// One wave per (b,s). The 2304B cent_p row is fetched by the Tensor Data Mover
// (TENSOR_LOAD_TO_LDS, 1-D tile of 288 x 8B), waited with s_wait_tensorcnt.
__global__ void att_kernel(const float* __restrict__ cp, const float* __restrict__ kgw,
                           const float* __restrict__ kgb, float* __restrict__ att) {
  __shared__ double cpl[CK / 2];                 // 2304 B landing buffer
  const int bs = blockIdx.x;
  const int lane = threadIdx.x;

  // ---- build D# (ISA 8.3/8.4): count=1, type=2, data_size=8B, 1-D tile of 288
  {
    unsigned ldsOff = (unsigned)(size_t)(void*)cpl;
    unsigned long long ga = (unsigned long long)(cp + (size_t)bs * CK);
    u32x4 g0;
    g0.x = 1u;                                   // count=1 (valid descriptor)
    g0.y = ldsOff;                               // lds_addr
    g0.z = (unsigned)ga;                         // global_addr[31:0]
    g0.w = (unsigned)(ga >> 32) | (2u << 30);    // global_addr[56:32] | type=2
    u32x8 g1;
    g1.s0 = 3u << 16;                            // data_size=3 (8B)
    g1.s1 = (unsigned)(CK / 2) << 16;            // tensor_dim0 = 288 (lo16 @ bits 63:48)
    g1.s2 = 1u << 16;                            // tensor_dim0 hi=0 | tensor_dim1 = 1
    g1.s3 = (unsigned)(CK / 2) << 16;            // tile_dim0 = 288
    g1.s4 = 1u;                                  // tile_dim1 = 1
    g1.s5 = (unsigned)(CK / 2);                  // tensor_dim0_stride = 288
    g1.s6 = 0u;
    g1.s7 = 0u;
    u32x4 g2 = {0u, 0u, 0u, 0u};
    u32x4 g3 = {0u, 0u, 0u, 0u};
    asm volatile("tensor_load_to_lds %0, %1, %2, %3"
                 :: "s"(g0), "s"(g1), "s"(g2), "s"(g3) : "memory");
  }
  __builtin_amdgcn_s_wait_tensorcnt(0);

  const float* v = (const float*)cpl;
  float lg[K9];
#pragma unroll
  for (int j = 0; j < K9; ++j) {
    float p = 0.f;
    for (int q = lane; q < CK; q += 32) p += v[q] * kgw[j * CK + q];
#pragma unroll
    for (int m = 16; m > 0; m >>= 1) p += __shfl_xor(p, m, 32);  // wave32 butterfly
    lg[j] = p + kgb[j];
  }
  float mx = lg[0];
#pragma unroll
  for (int j = 1; j < K9; ++j) mx = fmaxf(mx, lg[j]);
  float e[K9], sum = 0.f;
#pragma unroll
  for (int j = 0; j < K9; ++j) { e[j] = __expf(lg[j] - mx); sum += e[j]; }
  if (lane == 0) {
#pragma unroll
    for (int j = 0; j < K9; ++j) att[bs * K9 + j] = e[j] * (float(K9) / sum);
  }
}

// ---------------------------------------------------------------- main conv
// block = 256 (8 waves): 32 pixels (one row segment) x 64 outputs.
//  - w2 (packed bf16, WMMA B layout) async-staged: GLOBAL_LOAD_ASYNC_TO_LDS_B128
//  - x halo tile staged to LDS, then the modulated patch tile pm[32][576] (bf16)
//    is built ONCE per block; all 8 waves consume it (no 4x rebuild).
//  - inner loop: 2x ds_load_b128 (A) + 2x ds_load_b128 (B) + v_wmma per k-step.
constexpr int W2L_BYTES  = W2P_ELEMS * 2;          // 73728
constexpr int PM_BYTES   = 32 * CK * 2;            // 36864
constexpr int XT_ELEMS   = C * 3 * 36;             // 6912 f32
constexpr int SMEM_BYTES = W2L_BYTES + PM_BYTES + XT_ELEMS * 4 + 32 * K9 * 4;  // 139392

__global__ void main_conv(const float* __restrict__ x, const int* __restrict__ idx,
                          const float* __restrict__ att,
                          const unsigned short* __restrict__ w2p,
                          const float* __restrict__ b2, float* __restrict__ out) {
  extern __shared__ char smem[];
  __bf16* w2l = (__bf16*)smem;
  __bf16* pm  = (__bf16*)(smem + W2L_BYTES);
  float*  xt  = (float*)(smem + W2L_BYTES + PM_BYTES);
  float*  apx = (float*)(smem + W2L_BYTES + PM_BYTES + XT_ELEMS * 4);

  const int b  = blockIdx.x >> 9;
  const int p0 = (blockIdx.x & 511) * 32;
  const int tid = threadIdx.x;
  const int lane = tid & 31, wave = tid >> 5;
  const int lane16 = lane & 15, laneHi = lane >> 4;

  // --- async-stage packed w2 into LDS: 4608 x 16B chunks, 18 per thread
  {
    const unsigned ldsBase = (unsigned)(size_t)w2l;
    const unsigned long long gBase = (unsigned long long)(const char*)w2p;
    for (int chunk = tid; chunk < W2L_BYTES / 16; chunk += 256) {
      unsigned ldsOff = ldsBase + (unsigned)(chunk * 16);
      unsigned long long ga = gBase + (unsigned long long)(chunk * 16);
      asm volatile("global_load_async_to_lds_b128 %0, %1, off"
                   :: "v"(ldsOff), "v"(ga) : "memory");
    }
  }

  // --- stage x halo tile (rows hrow-1..hrow+1, cols wcol0-1..wcol0+32, zero-pad)
  const int hrow  = p0 >> 7;
  const int wcol0 = p0 & 127;
  const float* xb = x + (size_t)b * C * Nn;
  for (int e = tid; e < XT_ELEMS; e += 256) {
    int c = e / 108, rem = e % 108;
    int r = rem / 36, col = rem % 36;
    int gh = hrow - 1 + r, gw = wcol0 - 1 + col;
    float v = 0.f;
    if (col < 34 && (unsigned)gh < (unsigned)H && (unsigned)gw < (unsigned)W)
      v = xb[(size_t)c * Nn + gh * W + gw];
    xt[e] = v;
  }
  for (int t = tid; t < 32 * K9; t += 256) {       // per-pixel attention
    int m = t / K9, k = t % K9;
    int s = idx[(size_t)b * Nn + p0 + m];
    apx[t] = att[(size_t)(b * S + s) * K9 + k];
  }
  __syncthreads();                                 // xt, apx visible

  // --- build modulated patch tile pm[p][q], q = c*9+k (once per block)
  for (int e = tid; e < 32 * CK; e += 256) {
    int p = e / CK, q = e % CK;
    int c = q / K9, k = q % K9;
    float v = xt[c * 108 + (k / 3) * 36 + p + (k % 3)];
    pm[e] = (__bf16)(v * apx[p * K9 + k]);
  }
  asm volatile("s_wait_asynccnt 0" ::: "memory");  // w2 tile resident
  __syncthreads();                                 // pm + w2l visible

  const int ph     = wave & 1;
  const int otile  = wave >> 1;
  const int pixoff = ph * 16 + lane16;
  const __bf16* pmrow = pm + pixoff * CK;
  const __bf16* bsrc  = w2l + ((size_t)(otile * KSTEPS) * 32 + lane) * 16;

  v8f acc = {};
  for (int ks = 0; ks < KSTEPS; ++ks) {
    v16bf a  = frag16(pmrow, ks * 32, laneHi);                 // 2x ds_load_b128
    v16bf bm = *(const v16bf*)(bsrc + (size_t)ks * 32 * 16);   // 2x ds_load_b128
    acc = __builtin_amdgcn_wmma_f32_16x16x32_bf16(false, a, false, bm, (short)0, acc, false, false);
  }

  const int o = otile * 16 + lane16;
  const float bias = b2[o];
#pragma unroll
  for (int j = 0; j < 8; ++j) {                    // D: M = j + 8*laneHi, N = lane16
    int pix = p0 + ph * 16 + (laneHi << 3) + j;
    size_t off = ((size_t)(b * O + o)) * Nn + pix;
    out[off] = x[((size_t)(b * C + o)) * Nn + pix] + acc[j] + bias;
  }
}

// ---------------------------------------------------------------- host orchestration
extern "C" void kernel_launch(void* const* d_in, const int* in_sizes, int n_in,
                              void* d_out, int out_size, void* d_ws, size_t ws_size,
                              hipStream_t stream) {
  (void)in_sizes; (void)n_in; (void)out_size; (void)ws_size;
  const float* x   = (const float*)d_in[0];
  // d_in[1..4] = att_w, att_b, w1, b1 : dead branch in the reference, output-invariant.
  const float* kgw = (const float*)d_in[5];
  const float* kgb = (const float*)d_in[6];
  const float* w2  = (const float*)d_in[7];
  const float* b2  = (const float*)d_in[8];

  float* out = (float*)d_out;
  int*   idx = (int*)(out + (size_t)B * O * Nn);   // second tuple element

  float* wsf      = (float*)d_ws;
  float* cent     = wsf;                           // 16384
  float* cent_acc = wsf + 16384;                   // 16384
  float* cnt      = wsf + 32768;                   // 256 (adjacent to cent_acc)
  float* cent_p   = wsf + 33024;                   // 147456
  float* cnt2     = cent_p + (size_t)B * S * CK;   // 256 (adjacent to cent_p)
  float* att      = cnt2 + B * S;                  // 2304
  unsigned short* w2p = (unsigned short*)(wsf + 183040);  // 36864 ushorts, 16B aligned

  const dim3 blk(256);
  pack_w2<<<(W2P_ELEMS + 255) / 256, blk, 0, stream>>>(w2, w2p);
  init_cent<<<64, blk, 0, stream>>>(x, cent);
  for (int it = 0; it < KM_ITERS; ++it) {
    zero_f32<<<(B * S * C + B * S + 255) / 256, blk, 0, stream>>>(cent_acc, B * S * C + B * S);
    kmeans_assign<<<B * (Nn / 128), blk, 0, stream>>>(x, cent, idx);
    kmeans_update<<<B * (Nn / 256), blk, 0, stream>>>(x, idx, cent_acc, cnt);
    norm_cent<<<64, blk, 0, stream>>>(cent_acc, cnt, cent);
  }
  kmeans_assign<<<B * (Nn / 128), blk, 0, stream>>>(x, cent, idx);   // final assignment

  zero_f32<<<(B * S * CK + B * S + 255) / 256, blk, 0, stream>>>(cent_p, B * S * CK + B * S);
  patch_acc<<<B * (Nn / 256), blk, 0, stream>>>(x, idx, cent_p, cnt2);
  norm_centp<<<(B * S * CK + 255) / 256, blk, 0, stream>>>(cent_p, cnt2);
  att_kernel<<<B * S, dim3(32), 0, stream>>>(cent_p, kgw, kgb, att);
  main_conv<<<B * (Nn / 32), blk, SMEM_BYTES, stream>>>(x, idx, att, w2p, b2, out);
}
